// LSTM_31645319037527
// MI455X (gfx1250) — matrix operand order repaired and checked by reading, not compiled, not run on hardware
//
#include <hip/hip_runtime.h>
#include <stdint.h>

// ---------------- problem constants ----------------
#define B_  128
#define T_  1024
#define I_  256
#define H_  1024
#define H4_ 4096
#define O_  128

typedef __attribute__((ext_vector_type(16))) __bf16 v16bf;
typedef __attribute__((ext_vector_type(8)))  float  v8f;

union Frag32B { uint4 u4[2]; v16bf v; };     // 16 bf16 elements / lane
union CFrag   { v8f   v;     float f[8]; };  // 16x16 f32 accum tile / lane

// round-to-nearest-even f32 -> bf16 bits
__device__ __forceinline__ uint16_t f2bf(float x) {
  union { float f; uint32_t u; } c; c.f = x;
  uint32_t u = c.u;
  uint32_t r = (u + 0x7FFFu + ((u >> 16) & 1u)) >> 16;
  return (uint16_t)r;
}

__global__ void zero_ws_kernel(uint32_t* __restrict__ p, int n) {
  int i = blockIdx.x * blockDim.x + threadIdx.x;
  int s = gridDim.x * blockDim.x;
  for (; i < n; i += s) p[i] = 0u;
}

__global__ void cvt_bf16_kernel(const float* __restrict__ src,
                                uint16_t* __restrict__ dst, int n) {
  int i = blockIdx.x * blockDim.x + threadIdx.x;
  int s = gridDim.x * blockDim.x;
  for (; i < n; i += s) dst[i] = f2bf(src[i]);
}

// A fragment: 16x32 bf16 tile of row-major [rows, ld] matrix.
// lane L holds row (mbase + L%16); halves 0-7 hold K = k + (L/16)*8 .. +7,
// halves 8-15 hold K = k + 16 + (L/16)*8 .. +7  (ISA 16-bit A layout).
__device__ __forceinline__ void load_a(Frag32B& a, const uint16_t* __restrict__ base,
                                       long ld, int mbase, int k, int lrow, int lhi) {
  const uint16_t* p = base + (size_t)(mbase + lrow) * ld + k + lhi * 8;
  a.u4[0] = *(const uint4*)(p);
  a.u4[1] = *(const uint4*)(p + 16);
}

// B fragment: 32x16 tile from row-major weight [N, ld] (so B = W^T tile).
// lane L holds output column (nrow + L%16); halves 0-7 hold K = k + (L/16)*8 .. +7,
// halves 8-15 hold K = k + 16 + (L/16)*8 .. +7 (same as A).
__device__ __forceinline__ void load_b(Frag32B& b, const uint16_t* __restrict__ w,
                                       long ld, int nrow, int k, int lrow, int lhi) {
  const uint16_t* p = w + (size_t)(nrow + lrow) * ld + k + lhi * 8;
  b.u4[0] = *(const uint4*)(p);
  b.u4[1] = *(const uint4*)(p + 16);
}

// Fused LSTM step with 4-way split-K.
//   gates = xA @ wih^T + hprev @ whh^T + bih + bhh, then cell update.
// Tile decomposition: 8 m-tiles x 64 n-tiles (16x16 each), every tile computed
// by 4 waves splitting K; partials combined in LDS; split-0 wave applies the
// nonlinearities and updates c / h.  One wave owns all 4 gate sections of its
// (batch x hidden) patch, so i/f/g/o never round-trip through memory.
// Grid: 256 blocks x 256 threads = 2048 waves.
__global__ void __launch_bounds__(256)
lstm_step_kernel(const uint16_t* __restrict__ xA, long ldx, int Kx,
                 const uint16_t* __restrict__ wih,
                 const uint16_t* __restrict__ whh,
                 const float* __restrict__ bih,
                 const float* __restrict__ bhh,
                 const uint16_t* __restrict__ hprev,
                 float* __restrict__ cstate,
                 uint16_t* __restrict__ hnext) {
  const int tid  = threadIdx.x;
  const int lane = tid & 31;
  const int lrow = lane & 15;
  const int lhi  = lane >> 4;
  const int w    = tid >> 5;        // 8 waves / block
  const int split = w >> 1;         // 4-way K split
  const int mloc  = w & 1;          // 2 m-tiles / block
  const int nbase = (blockIdx.x >> 2) << 4;                 // 64 n-tiles
  const int mbase = (((blockIdx.x & 3) << 1) + mloc) << 4;  // 8 m-tiles

  CFrag acc[4];
#pragma unroll
  for (int g = 0; g < 4; ++g)
#pragma unroll
    for (int e = 0; e < 8; ++e) acc[g].f[e] = 0.0f;

  // ---- input-side projection: x_t @ wih^T  (K strided by split) ----
  for (int k = split * 32; k < Kx; k += 128) {
    Frag32B a; load_a(a, xA, ldx, mbase, k, lrow, lhi);
#pragma unroll
    for (int g = 0; g < 4; ++g) {
      Frag32B b; load_b(b, wih, Kx, g * H_ + nbase, k, lrow, lhi);
      acc[g].v = __builtin_amdgcn_wmma_f32_16x16x32_bf16(
          false, a.v, false, b.v, (short)0, acc[g].v, false, false);
    }
  }

  // ---- recurrent projection: h_{t-1} @ whh^T ----
  for (int k = split * 32; k < H_; k += 128) {
    Frag32B a; load_a(a, hprev, H_, mbase, k, lrow, lhi);
#pragma unroll
    for (int g = 0; g < 4; ++g) {
      Frag32B b; load_b(b, whh, H_, g * H_ + nbase, k, lrow, lhi);
      acc[g].v = __builtin_amdgcn_wmma_f32_16x16x32_bf16(
          false, a.v, false, b.v, (short)0, acc[g].v, false, false);
    }
  }

  // ---- combine split-K partials through LDS ----
  __shared__ float part[2][3][4][256];  // [m-tile][split-1][gate][lane*8+r]
  const int lin = lane * 8;
  if (split != 0) {
#pragma unroll
    for (int g = 0; g < 4; ++g)
#pragma unroll
      for (int r = 0; r < 8; ++r) part[mloc][split - 1][g][lin + r] = acc[g].f[r];
  }
  __syncthreads();
  if (split != 0) return;

#pragma unroll
  for (int g = 0; g < 4; ++g)
#pragma unroll
    for (int r = 0; r < 8; ++r)
      acc[g].f[r] += part[mloc][0][g][lin + r] +
                     part[mloc][1][g][lin + r] +
                     part[mloc][2][g][lin + r];

  // ---- biases + cell update (i, f, g, o) ----
  const int hcol = nbase + lrow;
  float bsum[4];
#pragma unroll
  for (int g = 0; g < 4; ++g) bsum[g] = bih[g * H_ + hcol] + bhh[g * H_ + hcol];

#pragma unroll
  for (int r = 0; r < 8; ++r) {
    const int brow = mbase + lhi * 8 + r;           // C layout: VGPR r -> M = r + 8*lhi
    const size_t idx = (size_t)brow * H_ + hcol;
    float iv = acc[0].f[r] + bsum[0];
    float fv = acc[1].f[r] + bsum[1];
    float gv = acc[2].f[r] + bsum[2];
    float ov = acc[3].f[r] + bsum[3];
    iv = 1.0f / (1.0f + __expf(-iv));
    fv = 1.0f / (1.0f + __expf(-fv));
    gv = tanhf(gv);
    ov = 1.0f / (1.0f + __expf(-ov));
    const float cn = fv * cstate[idx] + iv * gv;
    cstate[idx] = cn;
    hnext[idx] = f2bf(ov * tanhf(cn));
  }
}

// out[B, O] = h1 @ wout^T + bout.  64 waves: (B/16) x (O/16) tiles.
__global__ void __launch_bounds__(256)
out_gemm_kernel(const uint16_t* __restrict__ h1,
                const uint16_t* __restrict__ wout,
                const float* __restrict__ bout,
                float* __restrict__ out) {
  const int tid  = threadIdx.x;
  const int lane = tid & 31;
  const int lrow = lane & 15;
  const int lhi  = lane >> 4;
  const int wid  = blockIdx.x * (blockDim.x >> 5) + (tid >> 5);
  const int mbase = (wid & 7) << 4;
  const int nbase = (wid >> 3) << 4;

  CFrag acc;
#pragma unroll
  for (int e = 0; e < 8; ++e) acc.f[e] = 0.0f;

  for (int k = 0; k < H_; k += 32) {
    Frag32B a; load_a(a, h1,   H_, mbase, k, lrow, lhi);
    Frag32B b; load_b(b, wout, H_, nbase, k, lrow, lhi);
    acc.v = __builtin_amdgcn_wmma_f32_16x16x32_bf16(
        false, a.v, false, b.v, (short)0, acc.v, false, false);
  }

  const int col = nbase + lrow;
  const float bb = bout[col];
#pragma unroll
  for (int r = 0; r < 8; ++r) {
    const int brow = mbase + lhi * 8 + r;
    out[brow * O_ + col] = acc.f[r] + bb;
  }
}

// ---------------------------------------------------------------------------
extern "C" void kernel_launch(void* const* d_in, const int* in_sizes, int n_in,
                              void* d_out, int out_size, void* d_ws, size_t ws_size,
                              hipStream_t stream) {
  const float* x     = (const float*)d_in[0];
  const float* w_ih0 = (const float*)d_in[1];
  const float* w_hh0 = (const float*)d_in[2];
  const float* b_ih0 = (const float*)d_in[3];
  const float* b_hh0 = (const float*)d_in[4];
  const float* w_ih1 = (const float*)d_in[5];
  const float* w_hh1 = (const float*)d_in[6];
  const float* b_ih1 = (const float*)d_in[7];
  const float* b_hh1 = (const float*)d_in[8];
  const float* w_out = (const float*)d_in[9];
  const float* b_out = (const float*)d_in[10];
  float* out = (float*)d_out;

  uint8_t* ws = (uint8_t*)d_ws;
  size_t off = 0;
  auto carve = [&](size_t bytes) -> void* {
    void* p = ws + off;
    off = (off + bytes + 255) & ~(size_t)255;
    return p;
  };

  // --- state region first (zeroed once per call) ---
  float*    c0  = (float*)   carve((size_t)B_ * H_ * 4);
  float*    c1  = (float*)   carve((size_t)B_ * H_ * 4);
  uint16_t* h0a = (uint16_t*)carve((size_t)B_ * H_ * 2);
  uint16_t* h0b = (uint16_t*)carve((size_t)B_ * H_ * 2);
  uint16_t* h1a = (uint16_t*)carve((size_t)B_ * H_ * 2);
  uint16_t* h1b = (uint16_t*)carve((size_t)B_ * H_ * 2);
  const size_t state_bytes = off;

  // --- bf16 operand region (rewritten every call from d_in) ---
  uint16_t* xb    = (uint16_t*)carve((size_t)B_ * T_ * I_ * 2);
  uint16_t* wih0b = (uint16_t*)carve((size_t)H4_ * I_ * 2);
  uint16_t* whh0b = (uint16_t*)carve((size_t)H4_ * H_ * 2);
  uint16_t* wih1b = (uint16_t*)carve((size_t)H4_ * H_ * 2);
  uint16_t* whh1b = (uint16_t*)carve((size_t)H4_ * H_ * 2);
  uint16_t* woutb = (uint16_t*)carve((size_t)O_ * H_ * 2);
  (void)ws_size; (void)in_sizes; (void)n_in; (void)out_size;

  // zero c/h state
  {
    int n = (int)(state_bytes / 4);
    int blocks = (n + 255) / 256; if (blocks > 1024) blocks = 1024;
    zero_ws_kernel<<<blocks, 256, 0, stream>>>((uint32_t*)ws, n);
  }

  // fp32 -> bf16 conversions
  auto cvt = [&](const float* s, uint16_t* d, long n) {
    int blocks = (int)((n + 255) / 256);
    if (blocks > 2048) blocks = 2048;
    cvt_bf16_kernel<<<blocks, 256, 0, stream>>>(s, d, (int)n);
  };
  cvt(x,     xb,    (long)B_ * T_ * I_);
  cvt(w_ih0, wih0b, (long)H4_ * I_);
  cvt(w_hh0, whh0b, (long)H4_ * H_);
  cvt(w_ih1, wih1b, (long)H4_ * H_);
  cvt(w_hh1, whh1b, (long)H4_ * H_);
  cvt(w_out, woutb, (long)O_ * H_);

  // --- sequential recurrence: 2 fused kernels per timestep ---
  uint16_t* h0_cur = h0a; uint16_t* h0_nxt = h0b;
  uint16_t* h1_cur = h1a; uint16_t* h1_nxt = h1b;
  const dim3 grid(256), block(256);
  for (int t = 0; t < T_; ++t) {
    // layer 0: A = x_t (bf16 view into [B,T,I], row stride T*I)
    lstm_step_kernel<<<grid, block, 0, stream>>>(
        xb + (size_t)t * I_, (long)T_ * I_, I_,
        wih0b, whh0b, b_ih0, b_hh0, h0_cur, c0, h0_nxt);
    // layer 1: A = h0_new
    lstm_step_kernel<<<grid, block, 0, stream>>>(
        h0_nxt, (long)H_, H_,
        wih1b, whh1b, b_ih1, b_hh1, h1_cur, c1, h1_nxt);
    uint16_t* tmp;
    tmp = h0_cur; h0_cur = h0_nxt; h0_nxt = tmp;
    tmp = h1_cur; h1_cur = h1_nxt; h1_nxt = tmp;
  }

  // --- final projection: h1_T @ w_out^T + b_out ---
  out_gemm_kernel<<<dim3(8), block, 0, stream>>>(h1_cur, woutb, b_out, out);
}
